// FlowMatchingBlock_87703232184355
// MI455X (gfx1250) — compile-verified, matrix operand order checked by reading, not run on hardware
//
#include <hip/hip_runtime.h>
#include <math.h>

typedef __bf16 bf16;
typedef __attribute__((ext_vector_type(16))) __bf16 v16bf;
typedef __attribute__((ext_vector_type(8)))  float  v8f;

#define BM 128
#define BN 64
#define BK 64
#define SBUF (BM*BK + BN*BK)   // bf16 elements per LDS stage buffer (12288)

// ---- CDNA5 async global->LDS copy (ASYNCcnt) ------------------------------
__device__ __forceinline__ unsigned lds_off(const void* p) {
    // low 32 bits of the flat address of an LDS object == logical LDS address
    return (unsigned)(unsigned long long)p;
}
__device__ __forceinline__ void async_ld_b128(unsigned lds, const void* gaddr) {
    asm volatile("global_load_async_to_lds_b128 %0, %1, off"
                 :: "v"(lds), "v"(gaddr) : "memory");
}
__device__ __forceinline__ void wait_async0() {
    asm volatile("s_wait_asynccnt 0x0" ::: "memory");
}

__device__ __forceinline__ unsigned pack_bf16(float a, float b) {
    union { bf16 h[2]; unsigned u; } cv;
    cv.h[0] = (bf16)a; cv.h[1] = (bf16)b;
    return cv.u;
}

// ---------------------------------------------------------------------------
// GEMM: C[m,n] = epi( sum_k A(m,k)*B(k,n) )
//   B is always bf16 stored N x K row-major (k-contiguous per output column).
//   AMODE 0: A is bf16, M x K row-major (async-copied).
//   AMODE 1: A(m,k) = Afp32[k*lda + m] * uvec[k]  (Kexp path, manual staging).
//   EPI 0: acc + bias[n]    EPI 1: exp(alpha*acc)
//   EPI 2: gelu(acc+bias)   EPI 3: acc * vrow[m]
//   OUT 0: fp32 C[m*ldc+n]  OUT 1: bf16 C[m*ldc+n]  OUT 2: bf16 C[n*ldc+m]
// 256 threads = 8 waves (4x2); each wave: 32x32 tile = 2x2 WMMA accumulators.
// Double-buffered LDS, two barriers per BK=64 step, 8 WMMAs per wave per step.
// ---------------------------------------------------------------------------
template<int AMODE, int EPI, int OUT>
__global__ __launch_bounds__(256)
void gemm_wmma(const void* __restrict__ Ap, int lda, long sA,
               const bf16* __restrict__ Bp, int ldb, long sB,
               void* __restrict__ Cp, int ldc, long sC,
               int Kdim,
               const float* __restrict__ bias,
               const float* __restrict__ uvec, long sU,
               const float* __restrict__ vrow, long sV,
               float alpha)
{
    __shared__ __align__(128) bf16 smem[2 * SBUF];

    const int b  = blockIdx.z;
    const int m0 = blockIdx.y * BM;
    const int n0 = blockIdx.x * BN;
    const int t    = threadIdx.x;
    const int lane = t & 31;
    const int wm   = (t >> 5) >> 1;     // 0..3
    const int wn   = (t >> 5) & 1;      // 0..1
    const int lrow  = lane & 15;
    const int khalf = lane >> 4;

    const bf16*  Bb = Bp + (long)b * sB;
    const float* ub = uvec ? uvec + (long)b * sU : nullptr;
    const float* vb = vrow ? vrow + (long)b * sV : nullptr;

    v8f acc[2][2] = {};

    // ---- tile staging ----
    auto stage = [&](int buf, int k0) {
        const int ao = buf * SBUF;
        const int bo = buf * SBUF + BM * BK;
        if (AMODE == 0) {
            const bf16* Ab = (const bf16*)Ap + (long)b * sA;
            #pragma unroll
            for (int i = 0; i < 4; i++) {               // 1024 x 16B chunks
                const int c   = i * 256 + t;
                const int row = c >> 3;
                const int kc  = (c & 7) * 8;
                async_ld_b128(lds_off(&smem[ao + row * BK + kc]),
                              Ab + (long)(m0 + row) * lda + k0 + kc);
            }
        } else {
            // A(m,k) = Afp32[k*lda + m] * u[k]; pack bf16 pairs -> b32 LDS stores
            const float* Af = (const float*)Ap + (long)b * sA;
            const int kk = (t >> 3) * 2;                // 0,2,..,62
            const int mm = (t & 7) * 16;                // 0..112
            const float u0 = ub[k0 + kk], u1 = ub[k0 + kk + 1];
            const float* s0 = Af + (long)(k0 + kk)     * lda + m0 + mm;
            const float* s1 = Af + (long)(k0 + kk + 1) * lda + m0 + mm;
            #pragma unroll
            for (int i = 0; i < 16; i++) {
                *(unsigned*)&smem[ao + (mm + i) * BK + kk] =
                    pack_bf16(s0[i] * u0, s1[i] * u1);
            }
        }
        #pragma unroll
        for (int i = 0; i < 2; i++) {                   // 512 x 16B chunks
            const int c   = i * 256 + t;
            const int row = c >> 3;
            const int kc  = (c & 7) * 8;
            async_ld_b128(lds_off(&smem[bo + row * BK + kc]),
                          Bb + (long)(n0 + row) * ldb + k0 + kc);
        }
    };

    stage(0, 0);
    for (int k0 = 0; k0 < Kdim; k0 += BK) {
        const int buf = (k0 / BK) & 1;
        wait_async0();
        __syncthreads();
        if (k0 + BK < Kdim) stage(buf ^ 1, k0 + BK);

        const int ao = buf * SBUF;
        const int bo = buf * SBUF + BM * BK;
        #pragma unroll
        for (int s = 0; s < 2; s++) {                   // two 32-deep k-slices
            v16bf bfrag[2];
            #pragma unroll
            for (int j = 0; j < 2; j++)
                bfrag[j] = *reinterpret_cast<const v16bf*>(
                    &smem[bo + (wn*32 + j*16 + lrow) * BK + s*32 + khalf*16]);
            #pragma unroll
            for (int i = 0; i < 2; i++) {
                v16bf afrag = *reinterpret_cast<const v16bf*>(
                    &smem[ao + (wm*32 + i*16 + lrow) * BK + s*32 + khalf*16]);
                #pragma unroll
                for (int j = 0; j < 2; j++)
                    acc[i][j] = __builtin_amdgcn_wmma_f32_16x16x32_bf16(
                        false, afrag, false, bfrag[j], (short)0, acc[i][j],
                        false, false);
            }
        }
        __syncthreads();
    }

    // ---- epilogue + store ----
    // C/D layout: VGPR r, lane L -> m = r + 8*(L>>4), n = L&15
    const int rbase = 8 * khalf;
    #pragma unroll
    for (int i = 0; i < 2; i++) {
        const int rowb = m0 + wm*32 + i*16 + rbase;
        #pragma unroll
        for (int j = 0; j < 2; j++) {
            const int col = n0 + wn*32 + j*16 + lrow;
            #pragma unroll
            for (int r = 0; r < 8; r++) {
                const float va = acc[i][j][r];
                float o;
                if (EPI == 0)      o = va + (bias ? bias[col] : 0.0f);
                else if (EPI == 1) o = expf(va * alpha);
                else if (EPI == 2) { const float g = va + bias[col];
                                     o = 0.5f * g * (1.0f + erff(g * 0.70710678118654752f)); }
                else               o = va * vb[rowb + r];

                if (OUT == 0)
                    ((float*)Cp + (long)b * sC)[(long)(rowb + r) * ldc + col] = o;
                else if (OUT == 1)
                    ((bf16*)Cp + (long)b * sC)[(long)(rowb + r) * ldc + col] = (bf16)o;
                else
                    ((bf16*)Cp + (long)b * sC)[(long)col * ldc + rowb + r] = (bf16)o;
            }
        }
    }
}

// ---- fp32 -> bf16 flat conversion ----
__global__ __launch_bounds__(256)
void f32_to_bf16(const float* __restrict__ x, bf16* __restrict__ y)
{
    const long i = (long)blockIdx.x * 256 + threadIdx.x;
    y[i] = (bf16)x[i];
}

// ---- weight convert + transpose: W (K x N) fp32 -> Wt (N x K) bf16 ----
__global__ __launch_bounds__(256)
void wT_bf16(const float* __restrict__ W, bf16* __restrict__ Wt, int K, int N)
{
    const int n = blockIdx.y * 256 + threadIdx.x;
    const int k = blockIdx.x;
    Wt[(long)n * K + k] = (bf16)W[(long)k * N + n];
}

__global__ __launch_bounds__(256)
void init_uv(float* __restrict__ u, float* __restrict__ v, int N, int M, int B)
{
    const int i = blockIdx.x * 256 + threadIdx.x;
    if (i < B * N) u[i] = 1.0f / (float)N;
    if (i < B * M) v[i] = 1.0f / (float)M;
}

// u[b,n] = 1 / (sum_m Kexp[b,n,m] * v[b,m] + 1e-8)   (L2-resident row reduce)
__global__ __launch_bounds__(256)
void sinkhorn_u(const float* __restrict__ Kexp, const float* __restrict__ v,
                float* __restrict__ u, int N, int Mc)
{
    const int b = blockIdx.y;
    const int n = blockIdx.x;
    const float* row = Kexp + ((long)b * N + n) * Mc;
    const float* vb  = v + (long)b * Mc;
    float s = 0.0f;
    for (int m = threadIdx.x; m < Mc; m += 256) s += row[m] * vb[m];
    __shared__ float red[256];
    red[threadIdx.x] = s; __syncthreads();
    for (int w = 128; w > 0; w >>= 1) {
        if (threadIdx.x < w) red[threadIdx.x] += red[threadIdx.x + w];
        __syncthreads();
    }
    if (threadIdx.x == 0) u[(long)b * N + n] = 1.0f / (red[0] + 1e-8f);
}

// v[b,m] = 1 / (sum_n Kexp[b,n,m] * u[b,n] + 1e-8)   (64 cols/block, 4 slices)
__global__ __launch_bounds__(256)
void sinkhorn_v(const float* __restrict__ Kexp, const float* __restrict__ u,
                float* __restrict__ v, int N, int Mc)
{
    const int b   = blockIdx.y;
    const int c   = threadIdx.x & 63;
    const int sl  = threadIdx.x >> 6;
    const int col = blockIdx.x * 64 + c;
    const float* base = Kexp + (long)b * N * Mc + col;
    const float* ub   = u + (long)b * N;
    float s = 0.0f;
    for (int n = sl; n < N; n += 4) s += base[(long)n * Mc] * ub[n];
    __shared__ float red[4][64];
    red[sl][c] = s; __syncthreads();
    if (threadIdx.x < 64) {
        const float tot = red[0][threadIdx.x] + red[1][threadIdx.x]
                        + red[2][threadIdx.x] + red[3][threadIdx.x];
        v[(long)b * Mc + blockIdx.x * 64 + threadIdx.x] = 1.0f / (tot + 1e-8f);
    }
}

// out = LN(x + addend)*g + beta  (D==512); optional bf16 mirror for next GEMM
__global__ __launch_bounds__(256)
void ln_residual(const float* __restrict__ x, const float* __restrict__ addend,
                 const float* __restrict__ g, const float* __restrict__ beta,
                 float* __restrict__ out, bf16* __restrict__ out_bf, int D)
{
    const long row = blockIdx.x;
    const float* xr = x + row * D;
    const float* ar = addend + row * D;
    const int t = threadIdx.x;
    const float y0 = xr[t] + ar[t];
    const float y1 = xr[t + 256] + ar[t + 256];
    __shared__ float red[256];
    red[t] = y0 + y1; __syncthreads();
    for (int w = 128; w > 0; w >>= 1) { if (t < w) red[t] += red[t + w]; __syncthreads(); }
    const float mu = red[0] / (float)D;
    __syncthreads();
    const float d0 = y0 - mu, d1 = y1 - mu;
    red[t] = d0 * d0 + d1 * d1; __syncthreads();
    for (int w = 128; w > 0; w >>= 1) { if (t < w) red[t] += red[t + w]; __syncthreads(); }
    const float rstd = rsqrtf(red[0] / (float)D + 1e-5f);
    const float o0 = d0 * rstd * g[t]       + beta[t];
    const float o1 = d1 * rstd * g[t + 256] + beta[t + 256];
    if (out)    { out[row * D + t] = o0; out[row * D + t + 256] = o1; }
    if (out_bf) { out_bf[row * D + t] = (bf16)o0; out_bf[row * D + t + 256] = (bf16)o1; }
}

extern "C" void kernel_launch(void* const* d_in, const int* in_sizes, int n_in,
                              void* d_out, int out_size, void* d_ws, size_t ws_size,
                              hipStream_t stream)
{
    const float* source = (const float*)d_in[0];
    const float* target = (const float*)d_in[1];
    const float* Wq = (const float*)d_in[2];  const float* bq = (const float*)d_in[3];
    const float* Wk = (const float*)d_in[4];  const float* bk = (const float*)d_in[5];
    const float* Wv = (const float*)d_in[6];  const float* bv = (const float*)d_in[7];
    const float* Wo = (const float*)d_in[8];  const float* bo = (const float*)d_in[9];
    const float* W1 = (const float*)d_in[10]; const float* b1 = (const float*)d_in[11];
    const float* W2 = (const float*)d_in[12]; const float* b2 = (const float*)d_in[13];
    const float* g1 = (const float*)d_in[14]; const float* be1 = (const float*)d_in[15];
    const float* g2 = (const float*)d_in[16]; const float* be2 = (const float*)d_in[17];

    const int Bb = 8, N = 2048, M = 2048, D = 512, F = 2048;
    const long tokD = (long)M * D;      // 1,048,576 elems / batch
    const long tokF = (long)M * F;
    const long nm   = (long)N * M;

    // ---- workspace layout (aliased lifetimes) ----
    char* ws = (char*)d_ws;
    float* Kexp  = (float*)(ws);                          // [4..7]; reused below
    bf16*  hbf   = (bf16*)(ws);                           // [10..11] (67MB, in Kexp slot)
    float* fbuf  = (float*)(ws + 67108864UL);             // [11..12] (in Kexp slot)
    float* mo    = (float*)(ws + 134217728UL);            // [8..9]
    float* tbuf  = (float*)(ws + 167772160UL);            // [9..12]
    float* uvec  = (float*)(ws + 201326592UL);
    float* vvec  = (float*)(ws + 201392128UL);
    bf16*  srcbf = (bf16*)(ws + 201457664UL);             // [conv..3]
    bf16*  matbf = (bf16*)(ws + 201457664UL);             // [7..8]  (reuses srcbf)
    bf16*  tgtbf = (bf16*)(ws + 218234880UL);             // [conv..1]
    bf16*  tbf   = (bf16*)(ws + 218234880UL);             // [9..10] (reuses tgtbf)
    bf16*  Qbf   = (bf16*)(ws + 235012096UL);             // [1..4]
    bf16*  Kbf   = (bf16*)(ws + 251789312UL);             // [2..4]
    bf16*  Vtbf  = (bf16*)(ws + 268566528UL);             // [3..7] (D x N, transposed)
    bf16*  WqT   = (bf16*)(ws + 285343744UL);
    bf16*  WkT   = (bf16*)(ws + 285868032UL);
    bf16*  WvT   = (bf16*)(ws + 286392320UL);
    bf16*  WoT   = (bf16*)(ws + 286916608UL);
    bf16*  W1T   = (bf16*)(ws + 287440896UL);
    bf16*  W2T   = (bf16*)(ws + 289538048UL);

    const dim3 blk(256);
    const dim3 gProj(D / BN, M / BM, Bb);   // (8,16,8)
    const dim3 gCost(M / BN, N / BM, Bb);   // (32,16,8)
    const dim3 gF1  (F / BN, M / BM, Bb);   // (32,16,8)

    // ---- one-time bf16 conversions / weight transposes ----
    f32_to_bf16<<<(Bb * tokD) / 256, blk, 0, stream>>>(source, srcbf);
    f32_to_bf16<<<(Bb * tokD) / 256, blk, 0, stream>>>(target, tgtbf);
    wT_bf16<<<dim3(D, D / 256), blk, 0, stream>>>(Wq, WqT, D, D);
    wT_bf16<<<dim3(D, D / 256), blk, 0, stream>>>(Wk, WkT, D, D);
    wT_bf16<<<dim3(D, D / 256), blk, 0, stream>>>(Wv, WvT, D, D);
    wT_bf16<<<dim3(D, D / 256), blk, 0, stream>>>(Wo, WoT, D, D);
    wT_bf16<<<dim3(D, F / 256), blk, 0, stream>>>(W1, W1T, D, F);
    wT_bf16<<<dim3(F, D / 256), blk, 0, stream>>>(W2, W2T, F, D);

    // 1-3: projections (bf16 outputs; V written transposed D x N)
    gemm_wmma<0,0,1><<<gProj, blk, 0, stream>>>(tgtbf, D, tokD, WqT, D, 0,
        Qbf, D, tokD, D, bq, nullptr, 0, nullptr, 0, 0.f);
    gemm_wmma<0,0,1><<<gProj, blk, 0, stream>>>(srcbf, D, tokD, WkT, D, 0,
        Kbf, D, tokD, D, bk, nullptr, 0, nullptr, 0, 0.f);
    gemm_wmma<0,0,2><<<gProj, blk, 0, stream>>>(srcbf, D, tokD, WvT, D, 0,
        Vtbf, N, tokD, D, bv, nullptr, 0, nullptr, 0, 0.f);

    // 4: Kexp[b,n,m] = exp( (K_n . Q_m) / (sqrt(64)*8*0.05) ), alpha = 0.3125
    gemm_wmma<0,1,0><<<gCost, blk, 0, stream>>>(Kbf, D, tokD, Qbf, D, tokD,
        Kexp, M, nm, D, nullptr, nullptr, 0, nullptr, 0, 0.3125f);

    // 5-6: Sinkhorn (10 alternating scalings over L2-resident 134MB matrix)
    init_uv<<<(Bb * N + 255) / 256, blk, 0, stream>>>(uvec, vvec, N, M, Bb);
    for (int it = 0; it < 10; it++) {
        sinkhorn_u<<<dim3(N, Bb), blk, 0, stream>>>(Kexp, vvec, uvec, N, M);
        sinkhorn_v<<<dim3(M / 64, Bb), blk, 0, stream>>>(Kexp, uvec, vvec, N, M);
    }

    // 7: matched[b,m,d] = v[m] * sum_n (u[n]*Kexp[n,m]) * V[n,d]  -> bf16
    gemm_wmma<1,3,1><<<gProj, blk, 0, stream>>>(Kexp, M, nm, Vtbf, N, tokD,
        matbf, D, tokD, N, nullptr, uvec, N, vvec, M, 0.f);

    // 8: mo = matched@Wo + bo (fp32)
    gemm_wmma<0,0,0><<<gProj, blk, 0, stream>>>(matbf, D, tokD, WoT, D, 0,
        mo, D, tokD, D, bo, nullptr, 0, nullptr, 0, 0.f);

    // 9: t = LN(target + mo)*g1 + be1  (fp32 + bf16 mirror)
    ln_residual<<<Bb * M, blk, 0, stream>>>(target, mo, g1, be1, tbuf, tbf, D);

    // 10: h = gelu(t@W1 + b1)  (bf16)
    gemm_wmma<0,2,1><<<gF1, blk, 0, stream>>>(tbf, D, tokD, W1T, D, 0,
        hbf, F, tokF, D, b1, nullptr, 0, nullptr, 0, 0.f);

    // 11: f = h@W2 + b2  (fp32)
    gemm_wmma<0,0,0><<<gProj, blk, 0, stream>>>(hbf, F, tokF, W2T, F, 0,
        fbuf, D, tokD, F, b2, nullptr, 0, nullptr, 0, 0.f);

    // 12: out = LN(t + f)*g2 + be2
    ln_residual<<<Bb * M, blk, 0, stream>>>(tbuf, fbuf, g2, be2,
        (float*)d_out, nullptr, D);

    (void)in_sizes; (void)n_in; (void)out_size; (void)ws_size;
}
